// BiLSTM_CRF_62474594288156
// MI455X (gfx1250) — matrix operand order repaired and checked by reading, not compiled, not run on hardware
//
#include <hip/hip_runtime.h>
#include <hip/hip_bf16.h>
#include <math.h>

// ---------------- constants ----------------
#define Bb   128
#define Ss   256
#define Ww   20
#define WE   300
#define CE   50
#define CC   100
#define DIN  400        // WE + CC
#define KPAD 416        // DIN padded to multiple of 32
#define KT_IH 13        // KPAD/32
#define Hh   256        // per-direction hidden
#define KT_HH 8         // Hh/32
#define G4H  1024       // 4*H
#define Tt   12
#define NEGV (-10000.0f)
#define START_TAG 9
#define STOP_TAG  10
#define PAD_TAG   11

typedef __attribute__((ext_vector_type(16))) _Float16 v16h;
typedef __attribute__((ext_vector_type(8)))  float    v8f;

// ---------------- packed-fragment index helpers (CDNA5 wave32 WMMA layouts) ----
// A (16x32 f16) per-lane: lane = hi*16 + m (m = row%16); element e:
//   e<8  -> K = hi*8 + e          (K 0..7 / 8..15)
//   e>=8 -> K = 16 + hi*8 + (e-8) (K 16..23 / 24..31)
// Packed storage: pa[((mt*ktiles + kt)*32 + lane)*16 + e]  (32 B per lane, contiguous)
__device__ __host__ inline long a_pack_idx(long row, int k, int ktiles) {
  int mt = (int)(row >> 4), m = (int)(row & 15);
  int kt = k >> 5, kk = k & 31;
  int hi = (kk >> 3) & 1;
  int e  = (kk & 7) + ((kk >> 4) << 3);
  int lane = hi * 16 + m;
  return (((long)mt * ktiles + kt) * 32 + lane) * 16 + e;
}

// B (32x16 f16) per-lane: lane = hi*16 + n (n = col%16); element e -> K = hi*8 + (e&7) + (e>>3)*16.
// Packed storage: pb[((nt*ktiles + kt)*32 + lane)*16 + e]
// C/D (16x16 f32): lane = hi*16 + n; reg r -> row = r + hi*8.
// Packed storage: pc[(tile*32 + lane)*8 + r]   (32 B per lane, contiguous)

__device__ inline float sigmf(float x) { return 1.0f / (1.0f + __expf(-x)); }

// ---------------- weight packing into B-fragment layout ----------------
// Conceptual B = w^T (K x N), w is (N x cols) row-major; zero pad K to ktiles*32.
__global__ void pack_b_frag_kernel(const float* __restrict__ w, _Float16* __restrict__ pb,
                                   int N, int cols, int ktiles) {
  long i = (long)blockIdx.x * blockDim.x + threadIdx.x;
  long total = (long)(N >> 4) * ktiles * 32 * 16;
  if (i >= total) return;
  int e    = (int)(i & 15);
  int lane = (int)((i >> 4) & 31);
  long tl  = i >> 9;
  int kt   = (int)(tl % ktiles);
  int nt   = (int)(tl / ktiles);
  int n = lane & 15, kbase = (lane >> 4) * 8;
  int k = kt * 32 + kbase + (e & 7) + ((e >> 3) << 4);
  int col = nt * 16 + n;
  pb[i] = (k < cols) ? (_Float16)w[(long)col * cols + k] : (_Float16)0.0f;
}

__global__ void add_bias_kernel(const float* __restrict__ a, const float* __restrict__ b,
                                float* __restrict__ o, int n) {
  int i = blockIdx.x * blockDim.x + threadIdx.x;
  if (i < n) o[i] = a[i] + b[i];
}

// ---------------- embedding + char CNN -> packed A fragments ----------------
// One block per token (b*S+s).  A row index = s*B + b, K width KPAD, ktiles=13.
__global__ void __launch_bounds__(128)
embed_kernel(const int* __restrict__ word_idxs, const int* __restrict__ char_idxs,
             const float* __restrict__ word_table, const float* __restrict__ char_table,
             const float* __restrict__ conv_w, const float* __restrict__ conv_b,
             _Float16* __restrict__ pa) {
  __shared__ float ce[Ww][CE];
  int token = blockIdx.x;           // b*S + s
  int b = token >> 8, s = token & 255;
  long row = (long)s * Bb + b;
  int tid = threadIdx.x;
  for (int i = tid; i < Ww * CE; i += 128) {
    int w = i / CE, e = i % CE;
    ce[w][e] = char_table[(long)char_idxs[(long)token * Ww + w] * CE + e];
  }
  int wid = word_idxs[token];
  for (int e = tid; e < WE; e += 128)
    pa[a_pack_idx(row, e, KT_IH)] = (_Float16)word_table[(long)wid * WE + e];
  for (int e = DIN + tid; e < KPAD; e += 128)
    pa[a_pack_idx(row, e, KT_IH)] = (_Float16)0.0f;
  __syncthreads();
  if (tid < CC) {
    int cc = tid;
    float mx = -1e30f;
    for (int w = 0; w < Ww; ++w) {
      float acc = conv_b[cc];
#pragma unroll
      for (int k = 0; k < 3; ++k) {
        int ww = w + k - 1;
        if (ww < 0 || ww >= Ww) continue;
        const float* cw = conv_w + (long)cc * (CE * 3) + k;   // conv_w[cc][e][k]
        for (int e = 0; e < CE; ++e) acc += cw[e * 3] * ce[ww][e];
      }
      mx = fmaxf(mx, acc);
    }
    pa[a_pack_idx(row, WE + cc, KT_IH)] = (_Float16)fmaxf(mx, 0.0f);
  }
}

// ---------------- input-projection GEMM on packed fragments, 2x4 blocking ------
// C(M x N) = A(M x 32*ktiles) * B + bias ; everything fragment-packed.
// Each wave computes a 32x64 region: 2 A-frags x 4 B-frags -> 8 WMMA per k-tile.
__global__ void __launch_bounds__(256)
gemm_f16_bias(const _Float16* __restrict__ Ap, const _Float16* __restrict__ Bp,
              const float* __restrict__ bias, float* __restrict__ Cp,
              int M, int N, int ktiles) {
  int lane = threadIdx.x & 31;
  int wave = threadIdx.x >> 5;
  int ntiles = N >> 4;            // 16-col tiles
  int nsup   = ntiles >> 2;       // 64-col supertiles
  long sup   = (long)blockIdx.x * 8 + wave;
  long total = (long)(M >> 5) * nsup;
  if (sup >= total) return;                   // whole-wave early out, EXEC stays full
  int ms = (int)(sup / nsup), ns = (int)(sup % nsup);
  int mt0 = ms * 2, nt0 = ns * 4;
  v8f acc[2][4];
#pragma unroll
  for (int i = 0; i < 2; ++i)
#pragma unroll
    for (int j = 0; j < 4; ++j)
#pragma unroll
      for (int r = 0; r < 8; ++r) acc[i][j][r] = 0.0f;
  const _Float16* ap0 = Ap + (((long)mt0 * ktiles) * 32 + lane) * 16;
  const _Float16* bp0 = Bp + (((long)nt0 * ktiles) * 32 + lane) * 16;
  for (int kt = 0; kt < ktiles; ++kt) {
    v16h a[2], b[4];
#pragma unroll
    for (int i = 0; i < 2; ++i)
      a[i] = *(const v16h*)(ap0 + ((long)i * ktiles + kt) * 512);
#pragma unroll
    for (int j = 0; j < 4; ++j)
      b[j] = *(const v16h*)(bp0 + ((long)j * ktiles + kt) * 512);
#pragma unroll
    for (int i = 0; i < 2; ++i)
#pragma unroll
      for (int j = 0; j < 4; ++j)
        acc[i][j] = __builtin_amdgcn_wmma_f32_16x16x32_f16(false, a[i], false, b[j],
                                                           (short)0, acc[i][j], false, false);
  }
#pragma unroll
  for (int i = 0; i < 2; ++i)
#pragma unroll
    for (int j = 0; j < 4; ++j) {
      float bv = bias[(nt0 + j) * 16 + (lane & 15)];
      v8f outv;
#pragma unroll
      for (int r = 0; r < 8; ++r) outv[r] = acc[i][j][r] + bv;
      long tile = (long)(mt0 + i) * ntiles + (nt0 + j);
      *(v8f*)(Cp + (tile * 32 + lane) * 8) = outv;
    }
}

// ---------------- persistent bidirectional LSTM (1 block per direction) ----------------
// xp: packed C tiles, tile = (s*8+bt)*64 + (g*16+jt) ; whh: packed B fragments (ktiles=8)
// lo: packed h tiles, tile index = ((dir*S + s)*8 + bt)*16 + jt
__global__ void __launch_bounds__(1024)
lstm_persistent(const float* __restrict__ xp_f, const float* __restrict__ xp_b,
                const _Float16* __restrict__ whB_f, const _Float16* __restrict__ whB_b,
                float* __restrict__ lo) {
  __shared__ __attribute__((aligned(32))) _Float16 h_sh[Bb * Hh]; // A-frag packed, 64 KB
  const int dir = blockIdx.x;
  const float* xp = dir ? xp_b : xp_f;
  const _Float16* whB = dir ? whB_b : whB_f;
  int lane = threadIdx.x & 31, wave = threadIdx.x >> 5;
  int bt  = wave & 7;                          // row tile 0..7  (B = 8*16)
  int jt0 = (wave >> 3) << 2;                  // first of 4 col tiles (H = 16*16)

  for (int i = threadIdx.x; i < Bb * Hh; i += 1024) h_sh[i] = (_Float16)0.0f;
  v8f cst[4];
#pragma unroll
  for (int jj = 0; jj < 4; ++jj)
#pragma unroll
    for (int r = 0; r < 8; ++r) cst[jj][r] = 0.0f;
  __syncthreads();

  for (int step = 0; step < Ss; ++step) {
    int s = dir ? (Ss - 1 - step) : step;
    v8f acc[4][4];                             // [jtile][gate], gate order i,f,g,o
#pragma unroll
    for (int jj = 0; jj < 4; ++jj)
#pragma unroll
      for (int g = 0; g < 4; ++g) {
        long tile = ((long)s * 8 + bt) * 64 + (g * 16 + jt0 + jj);
        acc[jj][g] = *(const v8f*)(xp + (tile * 32 + lane) * 8);
      }
    for (int kt = 0; kt < KT_HH; ++kt) {
      v16h a = *(const v16h*)(&h_sh[(((long)bt * KT_HH + kt) * 32 + lane) * 16]);
#pragma unroll
      for (int jj = 0; jj < 4; ++jj)
#pragma unroll
        for (int g = 0; g < 4; ++g) {
          int nt = g * 16 + jt0 + jj;
          v16h b = *(const v16h*)(whB + (((long)nt * KT_HH + kt) * 32 + lane) * 16);
          acc[jj][g] = __builtin_amdgcn_wmma_f32_16x16x32_f16(false, a, false, b,
                                                              (short)0, acc[jj][g], false, false);
        }
    }
    float hnew[4][8];
#pragma unroll
    for (int jj = 0; jj < 4; ++jj) {
      v8f hv;
#pragma unroll
      for (int r = 0; r < 8; ++r) {
        float ig = sigmf(acc[jj][0][r]);
        float fg = sigmf(acc[jj][1][r]);
        float gg = tanhf(acc[jj][2][r]);
        float og = sigmf(acc[jj][3][r]);
        float c  = fg * cst[jj][r] + ig * gg;
        cst[jj][r] = c;
        float h = og * tanhf(c);
        hnew[jj][r] = h;
        hv[r] = h;
      }
      long tile = (((long)dir * Ss + s) * 8 + bt) * 16 + (jt0 + jj);
      *(v8f*)(lo + (tile * 32 + lane) * 8) = hv;   // packed h out (one 32 B store)
    }
    __syncthreads();                           // all reads of h_sh done
    {
      int m  = lane & 15, hi = lane >> 4;
#pragma unroll
      for (int jj = 0; jj < 4; ++jj)
#pragma unroll
        for (int r = 0; r < 8; ++r) {
          int brow = bt * 16 + r + hi * 8;
          int j    = (jt0 + jj) * 16 + m;
          h_sh[a_pack_idx(brow, j, KT_HH)] = (_Float16)hnew[jj][r];
        }
    }
    __syncthreads();
  }
}

// ---------------- tag projection (tiny: T=12), reads packed h ----------------
__device__ inline long lo_pack_idx(int dir, int s, int b, int j) {
  int bt = b >> 4, rr = b & 15, hi = rr >> 3, r = rr & 7;
  int jt = j >> 4, lanep = hi * 16 + (j & 15);
  return ((((long)dir * Ss + s) * 8 + bt) * 16 + jt) * 256 + lanep * 8 + r;
}

__global__ void __launch_bounds__(256)
proj_kernel(const float* __restrict__ lo, const float* __restrict__ w,
            const float* __restrict__ bias, float* __restrict__ feats) {
  __shared__ float w_sh[Tt * 2 * Hh];          // 24 KB
  for (int i = threadIdx.x; i < Tt * 2 * Hh; i += 256) w_sh[i] = w[i];
  __syncthreads();
  int token = blockIdx.x * 256 + threadIdx.x;  // b*S + s
  if (token >= Bb * Ss) return;
  int b = token >> 8, s = token & 255;
  float acc[Tt];
#pragma unroll
  for (int t = 0; t < Tt; ++t) acc[t] = bias[t];
  for (int dir = 0; dir < 2; ++dir)
    for (int j = 0; j < Hh; ++j) {
      float xv = lo[lo_pack_idx(dir, s, b, j)];
      int k = dir * Hh + j;
#pragma unroll
      for (int t = 0; t < Tt; ++t) acc[t] += xv * w_sh[t * (2 * Hh) + k];
    }
#pragma unroll
  for (int t = 0; t < Tt; ++t) feats[(long)token * Tt + t] = acc[t];
}

// ---------------- Viterbi + backtrace (one thread per batch row) ----------------
__global__ void __launch_bounds__(128)
viterbi_kernel(const float* __restrict__ feats, const int* __restrict__ mask,
               const float* __restrict__ trans, int* __restrict__ bp,
               float* __restrict__ out) {
  __shared__ float tr[Tt * Tt];
  int b = threadIdx.x;
  for (int i = b; i < Tt * Tt; i += 128) tr[i] = trans[i];
  __syncthreads();
  float alpha[Tt];
#pragma unroll
  for (int t = 0; t < Tt; ++t) alpha[t] = NEGV;
  alpha[START_TAG] = 0.0f;
  int len = 0;
  for (int s = 0; s < Ss; ++s) len += mask[(long)b * Ss + s];
  for (int s = 0; s < Ss; ++s) {
    int valid = mask[(long)b * Ss + s];
    float na[Tt];
#pragma unroll
    for (int t = 0; t < Tt; ++t) {
      float best = alpha[0] + tr[t];
      int arg = 0;
#pragma unroll
      for (int p = 1; p < Tt; ++p) {
        float v = alpha[p] + tr[p * Tt + t];
        if (v > best) { best = v; arg = p; }
      }
      bp[((long)s * Bb + b) * Tt + t] = arg;
      na[t] = best + feats[((long)b * Ss + s) * Tt + t];
    }
    if (valid)
#pragma unroll
      for (int t = 0; t < Tt; ++t) alpha[t] = na[t];
  }
  float bs = alpha[0] + tr[STOP_TAG * Tt + 0];
  int bl = 0;
#pragma unroll
  for (int t = 1; t < Tt; ++t) {
    float v = alpha[t] + tr[STOP_TAG * Tt + t];
    if (v > bs) { bs = v; bl = t; }
  }
  out[b] = bs;
  int tag = bl;
  for (int s = Ss - 1; s >= 0; --s) {
    int valid = (s < len) ? 1 : 0;
    out[Bb + (long)b * Ss + s] = (float)(valid ? tag : PAD_TAG);
    if (valid) tag = bp[((long)s * Bb + b) * Tt + tag];
  }
}

// ---------------- host-side orchestration ----------------
extern "C" void kernel_launch(void* const* d_in, const int* in_sizes, int n_in,
                              void* d_out, int out_size, void* d_ws, size_t ws_size,
                              hipStream_t stream) {
  const int*   word_idxs  = (const int*)d_in[0];
  const int*   char_idxs  = (const int*)d_in[1];
  const int*   mask       = (const int*)d_in[2];
  const float* word_table = (const float*)d_in[3];
  const float* char_table = (const float*)d_in[4];
  const float* conv_w     = (const float*)d_in[5];
  const float* conv_b     = (const float*)d_in[6];
  const float* w_ih_f     = (const float*)d_in[7];
  const float* w_hh_f     = (const float*)d_in[8];
  const float* b_ih_f     = (const float*)d_in[9];
  const float* b_hh_f     = (const float*)d_in[10];
  const float* w_ih_b     = (const float*)d_in[11];
  const float* w_hh_b     = (const float*)d_in[12];
  const float* b_ih_b     = (const float*)d_in[13];
  const float* b_hh_b     = (const float*)d_in[14];
  const float* h2t_w      = (const float*)d_in[15];
  const float* h2t_b      = (const float*)d_in[16];
  const float* trans      = (const float*)d_in[17];

  char* ws = (char*)d_ws;
  size_t off = 0;
  auto alloc = [&](size_t bytes) { size_t o = off; off = (off + bytes + 255) & ~(size_t)255; return o; };
  _Float16* embA   = (_Float16*)(ws + alloc((size_t)Bb * Ss * KPAD * 2));   // packed A frags
  _Float16* wihB_f = (_Float16*)(ws + alloc((size_t)KPAD * G4H * 2));       // packed B frags
  _Float16* wihB_b = (_Float16*)(ws + alloc((size_t)KPAD * G4H * 2));
  _Float16* whhB_f = (_Float16*)(ws + alloc((size_t)Hh * G4H * 2));
  _Float16* whhB_b = (_Float16*)(ws + alloc((size_t)Hh * G4H * 2));
  float*    bias_f = (float*)(ws + alloc((size_t)G4H * 4));
  float*    bias_b = (float*)(ws + alloc((size_t)G4H * 4));
  float*    xp_f   = (float*)(ws + alloc((size_t)Bb * Ss * G4H * 4));       // packed C tiles
  float*    xp_b   = (float*)(ws + alloc((size_t)Bb * Ss * G4H * 4));
  float*    lo     = (float*)(ws + alloc((size_t)Bb * Ss * 2 * Hh * 4));    // packed h tiles
  float*    feats  = (float*)(ws + alloc((size_t)Bb * Ss * Tt * 4));
  int*      bp     = (int*)(ws + alloc((size_t)Bb * Ss * Tt * 4));

  // 1) pack weights into B-fragment layout, combine biases
  {
    long totIh = (long)KPAD * G4H, totHh = (long)Hh * G4H;
    pack_b_frag_kernel<<<(int)((totIh + 255) / 256), 256, 0, stream>>>(w_ih_f, wihB_f, G4H, DIN, KT_IH);
    pack_b_frag_kernel<<<(int)((totIh + 255) / 256), 256, 0, stream>>>(w_ih_b, wihB_b, G4H, DIN, KT_IH);
    pack_b_frag_kernel<<<(int)((totHh + 255) / 256), 256, 0, stream>>>(w_hh_f, whhB_f, G4H, Hh, KT_HH);
    pack_b_frag_kernel<<<(int)((totHh + 255) / 256), 256, 0, stream>>>(w_hh_b, whhB_b, G4H, Hh, KT_HH);
    add_bias_kernel<<<(G4H + 255) / 256, 256, 0, stream>>>(b_ih_f, b_hh_f, bias_f, G4H);
    add_bias_kernel<<<(G4H + 255) / 256, 256, 0, stream>>>(b_ih_b, b_hh_b, bias_b, G4H);
  }
  // 2) embeddings + char CNN -> packed A fragments (rows ordered s*B+b)
  embed_kernel<<<Bb * Ss, 128, 0, stream>>>(word_idxs, char_idxs, word_table, char_table,
                                            conv_w, conv_b, embA);
  // 3) input projection GEMMs (WMMA, 2x4 blocked): xp = emb @ W_ih^T + (b_ih + b_hh)
  {
    int M = Bb * Ss, N = G4H;
    long sups = (long)(M / 32) * (N / 64);
    int blocks = (int)((sups + 7) / 8);
    gemm_f16_bias<<<blocks, 256, 0, stream>>>(embA, wihB_f, bias_f, xp_f, M, N, KT_IH);
    gemm_f16_bias<<<blocks, 256, 0, stream>>>(embA, wihB_b, bias_b, xp_b, M, N, KT_IH);
  }
  // 4) persistent recurrent LSTM, both directions in parallel
  lstm_persistent<<<2, 1024, 0, stream>>>(xp_f, xp_b, whhB_f, whhB_b, lo);
  // 5) tag projection
  proj_kernel<<<(Bb * Ss + 255) / 256, 256, 0, stream>>>(lo, h2t_w, h2t_b, feats);
  // 6) Viterbi decode + backtrace -> d_out = [scores(128) | paths(128*256) as float]
  viterbi_kernel<<<1, 128, 0, stream>>>(feats, mask, trans, bp, (float*)d_out);
}
